// SlotAttention_16269336117868
// MI455X (gfx1250) — compile-verified, matrix operand order checked by reading, not run on hardware
//
#include <hip/hip_runtime.h>
#include <hip/hip_bf16.h>
#include <math.h>

#define B_     64
#define N_     4096
#define C_     256
#define K_     8
#define ITERS_ 3
#define CHUNK_ 512
#define CH_    (N_ / CHUNK_)   // 8 key-chunks per batch

typedef __attribute__((ext_vector_type(16))) __bf16 bf16x16;
typedef __attribute__((ext_vector_type(8)))  __bf16 bf16x8;
typedef __attribute__((ext_vector_type(8)))  float  f32x8;
typedef int i32x4v __attribute__((vector_size(16)));   // for async-LDS builtin

// Probe for gfx1250 async global->LDS builtins (ASYNCcnt path).
#if defined(__has_builtin)
#if __has_builtin(__builtin_amdgcn_global_load_async_to_lds_b128) && \
    __has_builtin(__builtin_amdgcn_s_wait_asynccnt)
#define HAVE_ASYNC_LDS 1
#endif
#endif

#ifdef HAVE_ASYNC_LDS
typedef __attribute__((address_space(1))) i32x4v* as1_i32x4p;
typedef __attribute__((address_space(3))) i32x4v* as3_i32x4p;
#endif

// float -> bf16 (round-to-nearest-even) via bit manipulation
__device__ __forceinline__ unsigned short f2bf(float f) {
  unsigned u = __float_as_uint(f);
  u += 0x7FFFu + ((u >> 16) & 1u);
  return (unsigned short)(u >> 16);
}

// Load one lane's 16-element WMMA fragment (A or B, 16x16x32 bf16 layout).
// p must already point at base + row*stride + kstep*32 + 8*half.
// Elements 0..7 = K {0..7}+8h, elements 8..15 = K {16..23}+8h (ISA 7.12.2).
__device__ __forceinline__ bf16x16 load_frag(const unsigned short* p) {
  union { bf16x16 v; bf16x8 h[2]; } u;
  u.h[0] = *reinterpret_cast<const bf16x8*>(p);
  u.h[1] = *reinterpret_cast<const bf16x8*>(p + 16);
  return u.v;
}

__device__ __forceinline__ f32x8 wmma_bf16(bf16x16 a, bf16x16 b, f32x8 c) {
  return __builtin_amdgcn_wmma_f32_16x16x32_bf16(false, a, false, b,
                                                 (short)0, c, false, false);
}

// Wave-per-row LayerNorm over 8x256 fp32 LDS matrix. i = wave id (row),
// each lane owns 8 contiguous columns.
__device__ __forceinline__ void ln_row8(const float* src, float* dst,
                                        const float* gg, const float* bb,
                                        int i, int lane) {
  int cb = lane * 8;
  float v[8]; float s = 0.f, s2 = 0.f;
#pragma unroll
  for (int j = 0; j < 8; ++j) { v[j] = src[i*256 + cb + j]; s += v[j]; s2 += v[j]*v[j]; }
#pragma unroll
  for (int m = 1; m < 32; m <<= 1) { s += __shfl_xor(s, m, 32); s2 += __shfl_xor(s2, m, 32); }
  float mean = s * (1.f/256.f);
  float var  = s2 * (1.f/256.f) - mean*mean;
  float rstd = rsqrtf(var + 1e-5f);
#pragma unroll
  for (int j = 0; j < 8; ++j)
    dst[i*256 + cb + j] = (v[j]-mean)*rstd*gg[cb+j] + bb[cb+j];
}

// ---------------------------------------------------------------------------
// Kernel 0: convert+transpose Wk, Wv (fp32 [Cin][Cout]) -> bf16 [Cout][Cin]
// so that WMMA B-fragments load 16B contiguous per lane.
// ---------------------------------------------------------------------------
__global__ __launch_bounds__(256) void wconv_kernel(
    const float* __restrict__ Wk, const float* __restrict__ Wv,
    unsigned short* __restrict__ WkT, unsigned short* __restrict__ WvT) {
  int idx = blockIdx.x * 256 + threadIdx.x;      // 0 .. 2*65536-1
  int m = idx >> 16;
  int e = idx & 0xFFFF;
  int o = e >> 8, c = e & 255;
  const float* W = m ? Wv : Wk;
  unsigned short* WT = m ? WvT : WkT;
  WT[o*256 + c] = f2bf(W[c*256 + o]);
}

// ---------------------------------------------------------------------------
// Kernel 1: x = LN(inputs); k = x@Wk + bk -> kbuf bf16 [B][N][C]
//                           v = x@Wv + bv -> vT   bf16 [B][C][N] (transposed)
// One block = 16 tokens, 8 waves. WMMA 16x16x32 bf16, 8 k-steps per tile.
// ---------------------------------------------------------------------------
__global__ __launch_bounds__(256) void proj_kernel(
    const float* __restrict__ inp,
    const float* __restrict__ lng, const float* __restrict__ lnb,
    const unsigned short* __restrict__ WkT, const unsigned short* __restrict__ WvT,
    const float* __restrict__ bk, const float* __restrict__ bv,
    unsigned short* __restrict__ kbuf, unsigned short* __restrict__ vT) {
  __shared__ __align__(16) unsigned short xs[16 * 256];  // LN'd tokens (A matrix)
  __shared__ __align__(16) unsigned short kt[16 * 256];  // k tile staging

  int t = threadIdx.x, lane = t & 31, w = t >> 5;
  int nl = lane & 15, gh = lane >> 4;
  size_t tok0 = (size_t)blockIdx.x * 16;
  int b  = (int)(tok0 / N_);
  int n0 = (int)(tok0 % N_);

  // --- LayerNorm: wave w handles rows 2w, 2w+1; half-wave per row ---
  {
    int row = 2*w + gh;
    const float* xr = inp + (tok0 + row) * C_ + nl * 16;
    float v[16]; float s = 0.f, s2 = 0.f;
#pragma unroll
    for (int i = 0; i < 16; ++i) { v[i] = xr[i]; s += v[i]; s2 += v[i]*v[i]; }
#pragma unroll
    for (int m = 1; m < 16; m <<= 1) { s += __shfl_xor(s, m, 32); s2 += __shfl_xor(s2, m, 32); }
    float mean = s * (1.f/256.f);
    float var  = s2 * (1.f/256.f) - mean*mean;
    float rstd = rsqrtf(var + 1e-5f);
    int cb = nl * 16;
#pragma unroll
    for (int i = 0; i < 16; ++i)
      xs[row*256 + cb + i] = f2bf((v[i]-mean)*rstd*lng[cb+i] + lnb[cb+i]);
  }
  __syncthreads();

  // --- 32 output col-tiles (16 for k, 16 for v); wave w does 4 of them ---
#pragma unroll
  for (int which = 0; which < 4; ++which) {
    int tt  = 2*w + (which & 1);
    bool isV = (which >= 2);
    const unsigned short* WT = isV ? WvT : WkT;
    f32x8 acc = {0.f,0.f,0.f,0.f,0.f,0.f,0.f,0.f};
#pragma unroll
    for (int s = 0; s < 8; ++s) {
      bf16x16 a  = load_frag(xs + nl*256 + s*32 + 8*gh);
      bf16x16 bb = load_frag(WT + (size_t)(tt*16 + nl)*256 + s*32 + 8*gh);
      acc = wmma_bf16(a, bb, acc);
    }
    int col = tt*16 + nl;
    if (!isV) {
      float bias = bk[col];
#pragma unroll
      for (int r = 0; r < 8; ++r)
        kt[(r + 8*gh)*256 + col] = f2bf(acc[r] + bias);
    } else {
      float bias = bv[col];
      union { unsigned short s[8]; bf16x8 v; } tmp;
#pragma unroll
      for (int r = 0; r < 8; ++r) tmp.s[r] = f2bf(acc[r] + bias);
      // tokens (r + 8*gh) are contiguous in vT row [channel][token]
      unsigned short* dst = vT + ((size_t)b*C_ + col)*N_ + n0 + 8*gh;
      *reinterpret_cast<bf16x8*>(dst) = tmp.v;
    }
  }
  __syncthreads();

  // --- coalesced store of k tile: 16 rows x 512B ---
  {
    int row = t >> 4, seg = t & 15;
    const unsigned short* src = &kt[row*256 + seg*16];
    unsigned short* dst = kbuf + ((size_t)b*N_ + n0 + row)*C_ + seg*16;
    reinterpret_cast<uint4*>(dst)[0] = reinterpret_cast<const uint4*>(src)[0];
    reinterpret_cast<uint4*>(dst)[1] = reinterpret_cast<const uint4*>(src)[1];
  }
}

// ---------------------------------------------------------------------------
// Kernel 2: attention streaming pass for one iteration.
// Block = (batch b, key-chunk ch of 512 keys). Computes for its chunk:
//   dots = q@k^T * scale  -> softmax over slots (lane-local: slots 0..7 live
//   in lanes 0..15 of the WMMA D layout) -> p = softmax + EPS (bf16 in LDS)
//   updates_partial += p @ v   (WMMA, v^T fragments contiguous)
//   rowsum_partial[slot] += sum_keys p
// Renormalization over keys is folded into the update kernel as a row scale.
// q tile (8KB) is staged into LDS via async global->LDS copy (ASYNCcnt).
// ---------------------------------------------------------------------------
__global__ __launch_bounds__(256) void attn_kernel(
    const unsigned short* __restrict__ qbuf,   // [B][16][C] bf16 (rows 8..15 zero)
    const unsigned short* __restrict__ kbuf,   // [B][N][C]  bf16
    const unsigned short* __restrict__ vT,     // [B][C][N]  bf16
    float* __restrict__ updP,                  // [B][CH_][8][C]
    float* __restrict__ rsP) {                 // [B][CH_][8]
  __shared__ __align__(16) unsigned short qs[16 * 256];    // staged q tile
  __shared__ __align__(16) unsigned short pbuf[16 * 256];  // attn tile (A matrix)
  __shared__ float rsum[8];

  int t = threadIdx.x, lane = t & 31, w = t >> 5;
  int nl = lane & 15, gh = lane >> 4;
  int b  = blockIdx.x / CH_;
  int ch = blockIdx.x % CH_;

  const unsigned short* q  = qbuf + (size_t)b * 16 * C_;
  const unsigned short* kb = kbuf + (size_t)b * N_ * C_;
  const unsigned short* vb = vT   + (size_t)b * C_ * N_;

  // --- stage q (16x256 bf16 = 8KB) into LDS ---
#ifdef HAVE_ASYNC_LDS
  {
    // 256 lanes x 16B x 2 = 8KB via GLOBAL_LOAD_ASYNC_TO_LDS_B128 (ASYNCcnt)
    unsigned short* qnc = const_cast<unsigned short*>(q);
    __builtin_amdgcn_global_load_async_to_lds_b128(
        (as1_i32x4p)(qnc + t*8), (as3_i32x4p)(qs + t*8), 0, 0);
    __builtin_amdgcn_global_load_async_to_lds_b128(
        (as1_i32x4p)(qnc + 2048 + t*8), (as3_i32x4p)(qs + 2048 + t*8), 0, 0);
    __builtin_amdgcn_s_wait_asynccnt(0);
  }
#else
  {
    const uint4* src = reinterpret_cast<const uint4*>(q);
    uint4* dst = reinterpret_cast<uint4*>(qs);
    dst[t]       = src[t];
    dst[t + 256] = src[t + 256];
  }
#endif

  if (t < 8) rsum[t] = 0.f;
  float rs[8];
#pragma unroll
  for (int r = 0; r < 8; ++r) rs[r] = 0.f;
  f32x8 u0 = {0.f,0.f,0.f,0.f,0.f,0.f,0.f,0.f};
  f32x8 u1 = {0.f,0.f,0.f,0.f,0.f,0.f,0.f,0.f};
  const float scale = 0.0625f;   // C^-0.5

  for (int kb2 = 0; kb2 < CHUNK_/256; ++kb2) {
    int keyBase = ch * CHUNK_ + kb2 * 256;
    __syncthreads();   // pbuf reuse guard (also orders q staging / rsum init)

    // prefetch next 256-key block of k and v streams into cache hierarchy
    if (kb2 + 1 < CHUNK_/256) {
      __builtin_prefetch(kb + (size_t)(keyBase + 256 + t) * C_, 0, 1);
      __builtin_prefetch(vb + (size_t)t * N_ + keyBase + 256, 0, 1);
    }

    // --- phase A: wave w computes attn tiles lt = 2w, 2w+1 (16 keys each) ---
#pragma unroll
    for (int which = 0; which < 2; ++which) {
      int lt = 2*w + which;
      int keyTile = keyBase + lt*16;
      f32x8 acc = {0.f,0.f,0.f,0.f,0.f,0.f,0.f,0.f};
#pragma unroll
      for (int s = 0; s < 8; ++s) {
        bf16x16 a  = load_frag(qs + nl*256 + s*32 + 8*gh);
        bf16x16 bb = load_frag(kb + (size_t)(keyTile + nl)*C_ + s*32 + 8*gh);
        acc = wmma_bf16(a, bb, acc);
      }
      if (gh == 0) {  // lanes 0..15 hold slots 0..7 for key column nl
        float mx = -1e30f;
#pragma unroll
        for (int r = 0; r < 8; ++r) mx = fmaxf(mx, acc[r]*scale);
        float e[8]; float ssum = 0.f;
#pragma unroll
        for (int r = 0; r < 8; ++r) { e[r] = __expf(acc[r]*scale - mx); ssum += e[r]; }
        float inv = 1.f / ssum;
#pragma unroll
        for (int r = 0; r < 8; ++r) {
          float p = e[r]*inv + 1e-8f;
          rs[r] += p;
          pbuf[r*256 + lt*16 + nl] = f2bf(p);
        }
      } else {        // padding slots 8..15 -> zero
#pragma unroll
        for (int r = 0; r < 8; ++r)
          pbuf[(r + 8)*256 + lt*16 + nl] = 0;
      }
    }
    __syncthreads();

    // --- phase B: wave w accumulates update col-tiles ct = 2w, 2w+1 ---
#pragma unroll
    for (int s = 0; s < 8; ++s) {
      bf16x16 a  = load_frag(pbuf + nl*256 + s*32 + 8*gh);
      bf16x16 b0 = load_frag(vb + (size_t)(2*w*16 + nl)*N_ + keyBase + s*32 + 8*gh);
      u0 = wmma_bf16(a, b0, u0);
      bf16x16 b1 = load_frag(vb + (size_t)((2*w+1)*16 + nl)*N_ + keyBase + s*32 + 8*gh);
      u1 = wmma_bf16(a, b1, u1);
    }
  }

  // --- reduce row sums (ds_add_f32) and write partials ---
  if (gh == 0) {
#pragma unroll
    for (int r = 0; r < 8; ++r) atomicAdd(&rsum[r], rs[r]);
  }
  __syncthreads();
  if (t < 8) rsP[((size_t)b*CH_ + ch)*8 + t] = rsum[t];
  if (gh == 0) {
    float* up = updP + ((size_t)b*CH_ + ch) * 8 * C_;
#pragma unroll
    for (int r = 0; r < 8; ++r) up[r*C_ + 2*w*16 + nl] = u0[r];
#pragma unroll
    for (int r = 0; r < 8; ++r) up[r*C_ + (2*w+1)*16 + nl] = u1[r];
  }
}

// ---------------------------------------------------------------------------
// Kernel 3: slot update. One block per batch, 256 threads (wave w = slot row,
// lane = 8 contiguous channels).
//   phase 0           : init slots = mu + exp(log_sigma)*noise; emit q
//   phase 1..ITERS    : reduce partials -> renormalize -> GRU -> FF; emit q
//   phase == ITERS    : write final slots to d_out
// ---------------------------------------------------------------------------
__global__ __launch_bounds__(256) void update_kernel(
    int phase,
    const float* __restrict__ mu, const float* __restrict__ log_sigma,
    const float* __restrict__ noise,
    const float* __restrict__ ln_s_g, const float* __restrict__ ln_s_b,
    const float* __restrict__ ln_ff_g, const float* __restrict__ ln_ff_b,
    const float* __restrict__ Wq, const float* __restrict__ bq,
    const float* __restrict__ W_ih, const float* __restrict__ b_ih,
    const float* __restrict__ W_hh, const float* __restrict__ b_hh,
    const float* __restrict__ W1, const float* __restrict__ b1,
    const float* __restrict__ W2, const float* __restrict__ b2,
    const float* __restrict__ updP, const float* __restrict__ rsP,
    float* __restrict__ slotsBuf, unsigned short* __restrict__ qbuf,
    float* __restrict__ d_out) {
  __shared__ float sl [8*256];
  __shared__ float sl2[8*256];
  __shared__ float snb[8*256];
  __shared__ float h1b[8*256];
  __shared__ float upd[8*256];
  __shared__ float rinv[8];

  int t = threadIdx.x, lane = t & 31;
  int i = t >> 5;            // slot row (8 waves = 8 slots)
  int cb = lane * 8;         // this thread's 8 channels
  int b = blockIdx.x;
  float* slotG = slotsBuf + (size_t)b * K_ * C_;

  if (phase == 0) {
#pragma unroll
    for (int j = 0; j < 8; ++j) {
      int c = cb + j;
      float s0 = mu[c] + __expf(log_sigma[c]) * noise[i*C_ + c];
      sl[i*256 + c] = s0;
      slotG[i*C_ + c] = s0;
    }
  } else {
#pragma unroll
    for (int j = 0; j < 8; ++j)
      sl[i*256 + cb + j] = slotG[i*C_ + cb + j];
    if (t < 8) {
      float s = 0.f;
      for (int ch = 0; ch < CH_; ++ch) s += rsP[((size_t)b*CH_ + ch)*8 + t];
      rinv[t] = 1.f / s;
    }
    __syncthreads();
    // reduce update partials, apply key-renormalization
#pragma unroll
    for (int j = 0; j < 8; ++j) {
      int c = cb + j; float s = 0.f;
      for (int ch = 0; ch < CH_; ++ch)
        s += updP[((size_t)b*CH_ + ch)*8*C_ + i*C_ + c];
      upd[i*256 + c] = s * rinv[i];
    }
    __syncthreads();
    // --- GRU cell ---
#pragma unroll 2
    for (int j = 0; j < 8; ++j) {
      int c = cb + j;
      float gxr = b_ih[c], gxz = b_ih[256 + c], gxn = b_ih[512 + c];
      float ghr = b_hh[c], ghz = b_hh[256 + c], ghn = b_hh[512 + c];
      for (int cc = 0; cc < 256; ++cc) {
        float uv = upd[i*256 + cc], sv = sl[i*256 + cc];
        const float* wi = W_ih + (size_t)cc*768;
        const float* wh = W_hh + (size_t)cc*768;
        gxr += uv*wi[c];       ghr += sv*wh[c];
        gxz += uv*wi[256 + c]; ghz += sv*wh[256 + c];
        gxn += uv*wi[512 + c]; ghn += sv*wh[512 + c];
      }
      float r = 1.f / (1.f + __expf(-(gxr + ghr)));
      float z = 1.f / (1.f + __expf(-(gxz + ghz)));
      float n = tanhf(gxn + r*ghn);
      sl2[i*256 + c] = (1.f - z)*n + z*sl[i*256 + c];
    }
    __syncthreads();
    // --- FF: slots += relu(LN(slots)@W1 + b1)@W2 + b2 ---
    ln_row8(sl2, snb, ln_ff_g, ln_ff_b, i, lane);
    __syncthreads();
#pragma unroll 2
    for (int j = 0; j < 8; ++j) {
      int c = cb + j; float a = b1[c];
      for (int cc = 0; cc < 256; ++cc) a += snb[i*256 + cc] * W1[(size_t)cc*256 + c];
      h1b[i*256 + c] = fmaxf(a, 0.f);
    }
    __syncthreads();
#pragma unroll 2
    for (int j = 0; j < 8; ++j) {
      int c = cb + j; float a = b2[c];
      for (int cc = 0; cc < 256; ++cc) a += h1b[i*256 + cc] * W2[(size_t)cc*256 + c];
      float ns = sl2[i*256 + c] + a;
      sl[i*256 + c] = ns;
      slotG[i*C_ + c] = ns;
    }
  }
  __syncthreads();

  if (phase == ITERS_) {
#pragma unroll
    for (int j = 0; j < 8; ++j)
      d_out[(size_t)b*K_*C_ + i*C_ + cb + j] = sl[i*256 + cb + j];
  } else {
    // q = LN(slots)@Wq + bq, bf16, padded to 16 rows
    ln_row8(sl, snb, ln_s_g, ln_s_b, i, lane);
    __syncthreads();
    unsigned short* qg = qbuf + (size_t)b * 16 * C_;
#pragma unroll 2
    for (int j = 0; j < 8; ++j) {
      int c = cb + j; float a = bq[c];
      for (int cc = 0; cc < 256; ++cc) a += snb[i*256 + cc] * Wq[(size_t)cc*256 + c];
      qg[i*C_ + c] = f2bf(a);
    }
#pragma unroll
    for (int j = 0; j < 8; ++j)
      qg[(8 + i)*C_ + cb + j] = 0;   // zero padding rows 8..15
  }
}

// ---------------------------------------------------------------------------
extern "C" void kernel_launch(void* const* d_in, const int* in_sizes, int n_in,
                              void* d_out, int out_size, void* d_ws, size_t ws_size,
                              hipStream_t stream) {
  const float* inputs    = (const float*)d_in[0];
  const float* noise     = (const float*)d_in[1];
  const float* mu        = (const float*)d_in[2];
  const float* log_sigma = (const float*)d_in[3];
  const float* ln_in_g   = (const float*)d_in[4];
  const float* ln_in_b   = (const float*)d_in[5];
  const float* ln_s_g    = (const float*)d_in[6];
  const float* ln_s_b    = (const float*)d_in[7];
  const float* ln_ff_g   = (const float*)d_in[8];
  const float* ln_ff_b   = (const float*)d_in[9];
  const float* Wq   = (const float*)d_in[10];
  const float* bq   = (const float*)d_in[11];
  const float* Wk   = (const float*)d_in[12];
  const float* bk   = (const float*)d_in[13];
  const float* Wv   = (const float*)d_in[14];
  const float* bv   = (const float*)d_in[15];
  const float* W_ih = (const float*)d_in[16];
  const float* b_ih = (const float*)d_in[17];
  const float* W_hh = (const float*)d_in[18];
  const float* b_hh = (const float*)d_in[19];
  const float* W1   = (const float*)d_in[20];
  const float* b1   = (const float*)d_in[21];
  const float* W2   = (const float*)d_in[22];
  const float* b2   = (const float*)d_in[23];
  float* out = (float*)d_out;

  // workspace carve-up
  char* ws = (char*)d_ws;
  unsigned short* WkT = (unsigned short*)ws; ws += (size_t)C_*C_*2;
  unsigned short* WvT = (unsigned short*)ws; ws += (size_t)C_*C_*2;
  unsigned short* kbuf = (unsigned short*)ws; ws += (size_t)B_*N_*C_*2;
  unsigned short* vT   = (unsigned short*)ws; ws += (size_t)B_*C_*N_*2;
  unsigned short* qbuf = (unsigned short*)ws; ws += (size_t)B_*16*C_*2;
  float* slotsBuf = (float*)ws; ws += (size_t)B_*K_*C_*4;
  float* updP     = (float*)ws; ws += (size_t)B_*CH_*8*C_*4;
  float* rsP      = (float*)ws; ws += (size_t)B_*CH_*8*4;

  wconv_kernel<<<512, 256, 0, stream>>>(Wk, Wv, WkT, WvT);
  proj_kernel<<<(B_*N_)/16, 256, 0, stream>>>(inputs, ln_in_g, ln_in_b,
                                              WkT, WvT, bk, bv, kbuf, vT);
  update_kernel<<<B_, 256, 0, stream>>>(0, mu, log_sigma, noise,
      ln_s_g, ln_s_b, ln_ff_g, ln_ff_b, Wq, bq, W_ih, b_ih, W_hh, b_hh,
      W1, b1, W2, b2, updP, rsP, slotsBuf, qbuf, out);
  for (int it = 1; it <= ITERS_; ++it) {
    attn_kernel<<<B_*CH_, 256, 0, stream>>>(qbuf, kbuf, vT, updP, rsP);
    update_kernel<<<B_, 256, 0, stream>>>(it, mu, log_sigma, noise,
        ln_s_g, ln_s_b, ln_ff_g, ln_ff_b, Wq, bq, W_ih, b_ih, W_hh, b_hh,
        W1, b1, W2, b2, updP, rsP, slotsBuf, qbuf, out);
  }
}
